// EdgeRes_15152644620609
// MI455X (gfx1250) — matrix operand order repaired and checked by reading, not compiled, not run on hardware
//
#include <hip/hip_runtime.h>
#include <hip/hip_bf16.h>

#define B_   8
#define NPTS 1024
#define KNB  8

typedef __attribute__((ext_vector_type(16))) __bf16 v16bf;
typedef __attribute__((ext_vector_type(8)))  __bf16 v8bf;
typedef __attribute__((ext_vector_type(8)))  float  v8f;

// ---------------- fused cast + pad + transpose: F (b,Cin,n) fp32 -> XT (b,n,Cpad) bf16 ----
__global__ __launch_bounds__(256)
void castT_kernel(const float* __restrict__ F, __bf16* __restrict__ XT,
                  int Cin, int Cpad) {
    __shared__ float tile[32][33];
    int i0 = blockIdx.x * 32;
    int c0 = blockIdx.y * 32;
    int b  = blockIdx.z;
    int tx = threadIdx.x;          // 0..31
    int ty = threadIdx.y;          // 0..7
    const float* Fb = F + (size_t)b * Cin * NPTS;
#pragma unroll
    for (int k = 0; k < 4; ++k) {
        int c = c0 + ty + k * 8;
        tile[ty + k * 8][tx] = (c < Cin) ? Fb[(size_t)c * NPTS + i0 + tx] : 0.0f;
    }
    __syncthreads();
    __bf16* XTb = XT + (size_t)b * NPTS * Cpad;
#pragma unroll
    for (int k = 0; k < 4; ++k) {
        int i = i0 + ty + k * 8;
        XTb[(size_t)i * Cpad + c0 + tx] = (__bf16)tile[tx][ty + k * 8];
    }
}

__global__ __launch_bounds__(256)
void nrm_kernel(const float* __restrict__ F, float* __restrict__ nrm, int Cin) {
    int t = blockIdx.x * 256 + threadIdx.x;   // B_*NPTS threads
    int i = t % NPTS;
    int b = t / NPTS;
    const float* p = F + (size_t)b * Cin * NPTS + i;
    float s = 0.0f;
    for (int c = 0; c < Cin; ++c) {
        float v = p[(size_t)c * NPTS];
        s += v * v;
    }
    nrm[t] = s;
}

// wPrime = [wA ; wB - wA] in bf16, K padded to Cpad
__global__ __launch_bounds__(256)
void wprep_kernel(const float* __restrict__ w, __bf16* __restrict__ wp,
                  int O, int Cin, int Cpad, int total) {
    int t = blockIdx.x * 256 + threadIdx.x;
    if (t >= total) return;
    int k = t % Cpad;
    int m = t / Cpad;        // 0..2O-1
    float v = 0.0f;
    if (k < Cin) {
        if (m < O) v = w[(size_t)m * (2 * Cin) + k];
        else {
            const float* row = w + (size_t)(m - O) * (2 * Cin);
            v = row[Cin + k] - row[k];
        }
    }
    wp[t] = (__bf16)v;
}

__global__ __launch_bounds__(256)
void zero_kernel(float* __restrict__ p, int n) {
    int t = blockIdx.x * 256 + threadIdx.x;
    if (t < n) p[t] = 0.0f;
}

// ---------------- bf16 WMMA GEMM:  C[b] = A (M x K) * BT[b]^T  ----------------
// A: (M x K) row-major, K-contiguous.  BT: (N x K) row-major (i.e. B transposed),
// so both A and B WMMA fragments are contiguous 16B/32B global vector loads.
// No LDS, no barriers. grid: (M/128, N/64, batch); 8 waves = 4(M) x 2(N),
// each wave computes a 32x32 tile with 2x2 accumulators.
__global__ __launch_bounds__(256)
void wmma_gemm_bf16(const __bf16* __restrict__ A, const __bf16* __restrict__ BT,
                    float* __restrict__ C, int M, int N, int Kd,
                    unsigned long long aBatchStride) {
    const int lane = threadIdx.x & 31;
    const int wave = threadIdx.x >> 5;
    const int wm   = wave >> 1;                 // 0..3
    const int wn   = wave & 1;                  // 0..1
    const int m0   = blockIdx.x * 128 + wm * 32;
    const int n0   = blockIdx.y * 64  + wn * 32;
    const int bz   = blockIdx.z;

    const __bf16* Ab = A  + (size_t)aBatchStride * bz;
    const __bf16* Bb = BT + (size_t)bz * N * Kd;
    float*        Cb = C  + (size_t)bz * M * N;

    const int rsel = lane & 15;
    const int hs   = lane >> 4;                 // lane half select

    // A fragment rows: lanes 0-15 and 16-31 both cover M rows 0..15 of the tile,
    // with K chunks hs*16+0..7 and hs*16+8..15 (contiguous, matching B)
    const __bf16* arow0 = Ab + (size_t)(m0 + rsel) * Kd;
    const __bf16* arow1 = arow0 + (size_t)16 * Kd;
    // B fragment rows: lane<16 -> (N=rsel, K k0..k0+15); lane>=16 -> (N=rsel, K k0+16..31)
    const __bf16* brow0 = Bb + (size_t)(n0 + rsel) * Kd + hs * 16;
    const __bf16* brow1 = brow0 + (size_t)16 * Kd;

    v8f acc00 = {0.f,0.f,0.f,0.f,0.f,0.f,0.f,0.f};
    v8f acc01 = acc00, acc10 = acc00, acc11 = acc00;

    for (int k0 = 0; k0 < Kd; k0 += 32) {
        v8bf a0lo = *(const v8bf*)(arow0 + k0 + hs * 16);
        v8bf a0hi = *(const v8bf*)(arow0 + k0 + hs * 16 + 8);
        v8bf a1lo = *(const v8bf*)(arow1 + k0 + hs * 16);
        v8bf a1hi = *(const v8bf*)(arow1 + k0 + hs * 16 + 8);
        v16bf b0  = *(const v16bf*)(brow0 + k0);
        v16bf b1  = *(const v16bf*)(brow1 + k0);
        v16bf a0, a1;
#pragma unroll
        for (int q = 0; q < 8; ++q) {
            a0[q] = a0lo[q]; a0[8 + q] = a0hi[q];
            a1[q] = a1lo[q]; a1[8 + q] = a1hi[q];
        }
        acc00 = __builtin_amdgcn_wmma_f32_16x16x32_bf16(false, a0, false, b0, (short)0, acc00, false, false);
        acc01 = __builtin_amdgcn_wmma_f32_16x16x32_bf16(false, a0, false, b1, (short)0, acc01, false, false);
        acc10 = __builtin_amdgcn_wmma_f32_16x16x32_bf16(false, a1, false, b0, (short)0, acc10, false, false);
        acc11 = __builtin_amdgcn_wmma_f32_16x16x32_bf16(false, a1, false, b1, (short)0, acc11, false, false);
    }

    // C/D layout: VGPR r: lane<16 -> M=r, lane>=16 -> M=8+r; N column = rsel
    const int ncol = n0 + rsel;
    const int mb   = m0 + hs * 8;
#pragma unroll
    for (int r = 0; r < 8; ++r) {
        Cb[(size_t)(mb + r)      * N + ncol     ] = acc00[r];
        Cb[(size_t)(mb + r)      * N + ncol + 16] = acc01[r];
        Cb[(size_t)(mb + r + 16) * N + ncol     ] = acc10[r];
        Cb[(size_t)(mb + r + 16) * N + ncol + 16] = acc11[r];
    }
}

// ---------------- top-8 neighbors: wave per (b,i) row ----------------
__global__ __launch_bounds__(256)
void topk_kernel(const float* __restrict__ G, const float* __restrict__ nrm,
                 int* __restrict__ idx) {
    int gwave = (blockIdx.x * 256 + threadIdx.x) >> 5;   // row id, 0..B_*NPTS-1
    int lane  = threadIdx.x & 31;
    int b = gwave / NPTS, i = gwave % NPTS;
    const float* g  = G + ((size_t)b * NPTS + i) * NPTS;
    const float* nr = nrm + (size_t)b * NPTS;
    float ni = nr[i];

    float vals[32];
#pragma unroll
    for (int t = 0; t < 32; ++t) {
        int j = lane + 32 * t;
        vals[t] = 2.0f * g[j] - ni - nr[j];    // = -||xi-xj||^2
    }

    int* out = idx + ((size_t)b * NPTS + i) * KNB;
    for (int s = 0; s < KNB; ++s) {
        float bv = -3.4e38f; int bt = 0;
#pragma unroll
        for (int t = 0; t < 32; ++t)
            if (vals[t] > bv) { bv = vals[t]; bt = t; }
        int bj = lane + 32 * bt;
        for (int off = 16; off > 0; off >>= 1) {
            float ov = __shfl_xor(bv, off, 32);
            int   oj = __shfl_xor(bj, off, 32);
            if (ov > bv || (ov == bv && oj < bj)) { bv = ov; bj = oj; }
        }
        if ((bj & 31) == lane) {
            int kt = bj >> 5;
#pragma unroll
            for (int t = 0; t < 32; ++t)
                if (t == kt) vals[t] = -3.4e38f;
        }
        if (lane == 0) out[s] = bj;
    }
}

// ---------------- BN statistics over val = yA[idx] + yD ----------------
__global__ __launch_bounds__(256)
void bn_stats_kernel(const float* __restrict__ Y, const int* __restrict__ idx,
                     float* __restrict__ stats, int O) {
    int o   = blockIdx.x;
    int t   = blockIdx.y * 256 + threadIdx.x;   // (b,i) pair, 0..8191
    int b   = t >> 10, i = t & 1023;
    const float* yA = Y + ((size_t)b * 2 * O + o) * NPTS;
    const float* yD = yA + (size_t)O * NPTS;
    const int*   ix = idx + ((size_t)b * NPTS + i) * KNB;
    float d = yD[i];
    float s1 = 0.0f, s2 = 0.0f;
#pragma unroll
    for (int j = 0; j < KNB; ++j) {
        float v = yA[ix[j]] + d;
        s1 += v; s2 += v * v;
    }
    __shared__ float r1[256], r2[256];
    r1[threadIdx.x] = s1; r2[threadIdx.x] = s2;
    __syncthreads();
    for (int w = 128; w > 0; w >>= 1) {
        if (threadIdx.x < w) {
            r1[threadIdx.x] += r1[threadIdx.x + w];
            r2[threadIdx.x] += r2[threadIdx.x + w];
        }
        __syncthreads();
    }
    if (threadIdx.x == 0) {
        atomicAdd(&stats[2 * o + 0], r1[0]);
        atomicAdd(&stats[2 * o + 1], r2[0]);
    }
}

// ---------------- fused BN apply + ReLU + max over k ----------------
__global__ __launch_bounds__(256)
void bn_apply_max_kernel(const float* __restrict__ Y, const int* __restrict__ idx,
                         const float* __restrict__ stats,
                         const float* __restrict__ gam, const float* __restrict__ bet,
                         float* __restrict__ out, int O, int relu) {
    int t = blockIdx.x * 256 + threadIdx.x;    // over B_*O*NPTS
    int i = t % NPTS;
    int r = t / NPTS;
    int o = r % O;
    int b = r / O;
    const float cnt = (float)(B_ * NPTS * KNB);
    float mu  = stats[2 * o] / cnt;
    float var = stats[2 * o + 1] / cnt - mu * mu;
    float sc  = gam[o] * rsqrtf(var + 1e-5f);
    float sh  = bet[o] - mu * sc;
    const float* yA = Y + ((size_t)b * 2 * O + o) * NPTS;
    float d = yA[(size_t)O * NPTS + i];
    const int* ix = idx + ((size_t)b * NPTS + i) * KNB;
    float m = -3.4e38f;
#pragma unroll
    for (int j = 0; j < KNB; ++j) {
        float v = (yA[ix[j]] + d) * sc + sh;
        if (relu) v = fmaxf(v, 0.0f);
        m = fmaxf(m, v);
    }
    out[((size_t)b * O + o) * NPTS + i] = m;
}

// ---------------- global max over points + concat skip ----------------
__global__ __launch_bounds__(256)
void globmax_kernel(const float* __restrict__ h3, float* __restrict__ glob) {
    int gwave = (blockIdx.x * 256 + threadIdx.x) >> 5;   // b*1024 rows
    int lane  = threadIdx.x & 31;
    int b = gwave >> 10, o = gwave & 1023;
    const float* p = h3 + ((size_t)b * 1024 + o) * NPTS;
    float m = -3.4e38f;
    for (int j = lane; j < NPTS; j += 32) m = fmaxf(m, p[j]);
    for (int off = 16; off > 0; off >>= 1) m = fmaxf(m, __shfl_xor(m, off, 32));
    if (lane == 0) glob[b * 1024 + o] = m;
}

__global__ __launch_bounds__(256)
void concat_kernel(const float* __restrict__ glob, const float* __restrict__ h1,
                   float* __restrict__ hcat) {
    size_t t = (size_t)blockIdx.x * 256 + threadIdx.x;   // B_*1088*NPTS
    int i = (int)(t % NPTS);
    size_t r = t / NPTS;
    int c = (int)(r % 1088);
    int b = (int)(r / 1088);
    float v = (c < 1024) ? glob[b * 1024 + c]
                         : h1[((size_t)b * 64 + (c - 1024)) * NPTS + i];
    hcat[t] = v;
}

// ---------------- launch ----------------
extern "C" void kernel_launch(void* const* d_in, const int* in_sizes, int n_in,
                              void* d_out, int out_size, void* d_ws, size_t ws_size,
                              hipStream_t stream) {
    const float* x = (const float*)d_in[0];
    const float *W[6], *Gm[6], *Bt[6];
    for (int l = 0; l < 6; ++l) {
        W[l]  = (const float*)d_in[1 + 3 * l];
        Gm[l] = (const float*)d_in[2 + 3 * l];
        Bt[l] = (const float*)d_in[3 + 3 * l];
    }

    // workspace carve (requires ~230 MB)
    char* wsp = (char*)d_ws;
    auto carve = [&](size_t bytes) -> void* {
        void* p = (void*)wsp;
        wsp += (bytes + 255) & ~(size_t)255;
        return p;
    };
    const int CMAX = 1088, OMAX = 1024, MMAX = 2048;
    __bf16* XTbf = (__bf16*)carve((size_t)B_ * NPTS * CMAX * 2);
    float*  Gbuf = (float*) carve((size_t)B_ * NPTS * NPTS * 4);
    float*  nrmb = (float*) carve((size_t)B_ * NPTS * 4);
    int*    idxb = (int*)   carve((size_t)B_ * NPTS * KNB * 4);
    float*  Ybuf = (float*) carve((size_t)B_ * MMAX * NPTS * 4);
    float*  h1b  = (float*) carve((size_t)B_ * 64 * NPTS * 4);
    float*  hAb  = (float*) carve((size_t)B_ * OMAX * NPTS * 4);
    float*  hBb  = (float*) carve((size_t)B_ * OMAX * NPTS * 4);
    float*  hcat = (float*) carve((size_t)B_ * 1088 * NPTS * 4);
    float*  glob = (float*) carve((size_t)B_ * 1024 * 4);
    float*  stats= (float*) carve((size_t)2 * OMAX * 4);
    __bf16* wp   = (__bf16*)carve((size_t)MMAX * CMAX * 2);

    auto run_layer = [&](const float* feat, int Cin, int Cpad, int O,
                         const float* w, const float* g, const float* bt,
                         float* out, int relu) {
        // fused cast+pad+transpose: feat (b,Cin,n) -> XT (b,n,Cpad) bf16
        castT_kernel<<<dim3(NPTS / 32, Cpad / 32, B_), dim3(32, 8), 0, stream>>>(
            feat, XTbf, Cin, Cpad);
        nrm_kernel<<<(B_ * NPTS) / 256, 256, 0, stream>>>(feat, nrmb, Cin);
        // Gram: G[b] = XT (n x Cpad) * XT^T  (both operands from XT)
        wmma_gemm_bf16<<<dim3(NPTS / 128, NPTS / 64, B_), 256, 0, stream>>>(
            XTbf, XTbf, Gbuf, NPTS, NPTS, Cpad, (unsigned long long)NPTS * Cpad);
        topk_kernel<<<(B_ * NPTS * 32) / 256, 256, 0, stream>>>(Gbuf, nrmb, idxb);
        // Y[b] = [wA ; wB-wA] (2O x Cpad) * XT^T
        int M = 2 * O;
        int totW = M * Cpad;
        wprep_kernel<<<(totW + 255) / 256, 256, 0, stream>>>(w, wp, O, Cin, Cpad, totW);
        wmma_gemm_bf16<<<dim3(M / 128, NPTS / 64, B_), 256, 0, stream>>>(
            wp, XTbf, Ybuf, M, NPTS, Cpad, 0ULL);
        zero_kernel<<<(2 * O + 255) / 256, 256, 0, stream>>>(stats, 2 * O);
        bn_stats_kernel<<<dim3(O, (B_ * NPTS) / 256), 256, 0, stream>>>(Ybuf, idxb, stats, O);
        bn_apply_max_kernel<<<(B_ * O * NPTS) / 256, 256, 0, stream>>>(
            Ybuf, idxb, stats, g, bt, out, O, relu);
    };

    run_layer(x,    3,    32,   64,  W[0], Gm[0], Bt[0], h1b, 1);          // h1
    run_layer(h1b,  64,   64,   128, W[1], Gm[1], Bt[1], hAb, 1);          // h2
    run_layer(hAb,  128,  128, 1024, W[2], Gm[2], Bt[2], hBb, 0);          // h3 (no relu)
    globmax_kernel<<<(B_ * 1024 * 32) / 256, 256, 0, stream>>>(hBb, glob);
    concat_kernel<<<(B_ * 1088 * NPTS) / 256, 256, 0, stream>>>(glob, h1b, hcat);
    run_layer(hcat, 1088, 1088, 512, W[3], Gm[3], Bt[3], hAb, 1);
    run_layer(hAb,  512,  512,  256, W[4], Gm[4], Bt[4], hBb, 1);
    run_layer(hBb,  256,  256,  128, W[5], Gm[5], Bt[5], (float*)d_out, 1);
}
